// ExemplarModel_51513837748520
// MI455X (gfx1250) — matrix operand order repaired and checked by reading, not compiled, not run on hardware
//
#include <hip/hip_runtime.h>
#include <hip/hip_bf16.h>

// ---------------------------------------------------------------------------
// ExemplarModel on MI455X (gfx1250):
//   cross = (x*Sigma) @ e^T  -> v_wmma_f32_16x16x32_bf16 (f32 accumulate)
//   dist  = x_sq + e_sq - 2*cross ; att = exp(-beta*dist)
//   logits = segment_sum(att by label) ; out = softmax(gamma*logits)
//
// Blocking rationale (23.3 TB/s HBM, 192 MB L2, 320 KB LDS/WGP):
//  * bf16 B matrix = 50 MB -> L2 resident after one conversion pass.
//  * 64x256 block tile, A+B staged per-block into LDS by the Tensor Data
//    Mover (tensor_load_to_lds, TENSORcnt, double buffered) -> B pulled from
//    L2 only once per 64 rows (~800 MB total instead of ~3.2 GB).
//  * D# pad feature (4 DWORDs per 16) gives 80B LDS row stride -> all 32
//    lanes' ds_load_b128 fragment reads start on distinct 4-bank boundaries
//    (conflict-free on 64 banks).
// Workspace requirement: ~53 MB.
// ---------------------------------------------------------------------------

typedef __bf16 bf16_t;
typedef __attribute__((ext_vector_type(16))) __bf16 v16bf;
typedef __attribute__((ext_vector_type(8)))  float  v8f;
typedef __attribute__((ext_vector_type(4)))  unsigned int u32x4;
typedef __attribute__((ext_vector_type(8)))  int   i32x8;
typedef __attribute__((ext_vector_type(4)))  int   i32x4;

#define NDIM 1024
#define DDIM 512
#define MDIM 50000
#define MPAD 50176   /* 196 * 256 */
#define NCLS 10
#define KCHUNK 32
#define NCHUNKS (DDIM / KCHUNK)   /* 16 */
#define ROWB 80                    /* padded LDS row stride: 64B data + 16B pad */

union Frag { v16bf v; uint4 q[2]; };
static_assert(sizeof(Frag) == 32, "frag size");

// ---------------- TDM: 2D tile (tile_d1 rows x 32 bf16) -> LDS ----------------
// data_size=2B; pad_enable, pad_interval=3 (16 DWORDs), pad_amount=3 (4 DWORDs)
// => LDS rows written with 80B stride.
__device__ __forceinline__ void tdm_load_2d(unsigned lds_off, const void* gaddr,
                                            unsigned tensor_d0, unsigned tensor_d1,
                                            unsigned stride0,
                                            unsigned tile_d0, unsigned tile_d1) {
    unsigned long long ga = (unsigned long long)gaddr;
    u32x4 g0;
    g0.x = 1u;                                      // count=1 valid descriptor
    g0.y = lds_off;                                 // lds_addr (bytes)
    g0.z = (unsigned)(ga & 0xFFFFFFFFu);            // global_addr[31:0]
    g0.w = (unsigned)((ga >> 32) & 0x01FFFFFFu)     // global_addr[56:32]
           | (2u << 30);                            // type = 2 ("image")
    i32x8 g1;
    g1[0] = (int)((1u << 16)                        // data_size = 2 bytes
                | (1u << 20)                        // pad_enable
                | (3u << 22)                        // pad_interval: 16 DWORDs
                | (3u << 25));                      // pad_amount:   4 DWORDs
    g1[1] = (int)(tensor_d0 << 16);                 // tensor_dim0[15:0] @ [63:48]
    g1[2] = (int)((tensor_d0 >> 16) | (tensor_d1 << 16));
    g1[3] = (int)((tensor_d1 >> 16) | (tile_d0 << 16));
    g1[4] = (int)tile_d1;                           // tile_dim1 (tile_dim2 = 0)
    g1[5] = (int)stride0;                           // tensor_dim0_stride[31:0]
    g1[6] = 0;
    g1[7] = 0;
    i32x4 z4 = {0, 0, 0, 0};
#if defined(__clang_major__) && (__clang_major__ >= 23)
    i32x8 z8 = {0, 0, 0, 0, 0, 0, 0, 0};
    __builtin_amdgcn_tensor_load_to_lds(g0, g1, z4, z4, z8, 0);
#else
    __builtin_amdgcn_tensor_load_to_lds(g0, g1, z4, z4, 0);
#endif
}

// ---------------- prep: xs = bf16(x * Sigma), x_sq[row] ----------------
__global__ void prep_x_kernel(const float* __restrict__ x,
                              const float* __restrict__ sig,
                              bf16_t* __restrict__ xs,
                              float* __restrict__ xsq) {
    const int row = blockIdx.x;
    const int t   = threadIdx.x;
    float partial = 0.f;
    for (int d = t; d < DDIM; d += 256) {
        float xv = x[(size_t)row * DDIM + d];
        float sv = sig[d];
        xs[(size_t)row * DDIM + d] = (bf16_t)(xv * sv);
        partial += xv * xv * sv;
    }
    __shared__ float red[256];
    red[t] = partial;
    __syncthreads();
    for (int s = 128; s > 0; s >>= 1) {
        if (t < s) red[t] += red[t + s];
        __syncthreads();
    }
    if (t == 0) xsq[row] = red[0];
}

// ---------------- prep: eb = bf16(e) (zero-padded), e_sq[row] ----------------
__global__ void prep_e_kernel(const float* __restrict__ e,
                              const float* __restrict__ sig,
                              bf16_t* __restrict__ eb,
                              float* __restrict__ esq) {
    const int row = blockIdx.x;
    const int t   = threadIdx.x;
    float partial = 0.f;
    if (row < MDIM) {
        for (int d = t; d < DDIM; d += 256) {
            float ev = e[(size_t)row * DDIM + d];
            eb[(size_t)row * DDIM + d] = (bf16_t)ev;
            partial += ev * ev * sig[d];
        }
    } else {
        for (int d = t; d < DDIM; d += 256)
            eb[(size_t)row * DDIM + d] = (bf16_t)0.f;
    }
    __shared__ float red[256];
    red[t] = partial;
    __syncthreads();
    for (int s = 128; s > 0; s >>= 1) {
        if (t < s) red[t] += red[t + s];
        __syncthreads();
    }
    if (t == 0) esq[row] = red[0];
}

__global__ void zero_kernel(float* __restrict__ p, int n) {
    int i = blockIdx.x * blockDim.x + threadIdx.x;
    if (i < n) p[i] = 0.f;
}

// ---------------- main: TDM-staged WMMA GEMM + exp + segment-sum ----------------
// grid = (MPAD/256, NDIM/64), block = 256 threads (8 waves).
// Block tile 64 rows x 256 cols; wave (wr,wc) owns 32 rows x 64 cols
// (2 A-frags x 4 B-tiles = 8 WMMA per K-chunk).
__global__ __launch_bounds__(256) void exemplar_main_kernel(
    const bf16_t* __restrict__ xs, const bf16_t* __restrict__ eb,
    const float* __restrict__ xsq, const float* __restrict__ esq,
    const int* __restrict__ labels, const float* __restrict__ beta_p,
    float* __restrict__ logits) {

    __shared__ __align__(16) unsigned char ldsA[2][64][ROWB];    // 10.0 KB
    __shared__ __align__(16) unsigned char ldsB[2][256][ROWB];   // 40.0 KB
    __shared__ float lgt[64][NCLS];                              //  2.5 KB

    const int lane  = threadIdx.x & 31;
    const int wid   = threadIdx.x >> 5;
    const int wr    = wid >> 2;         // 0..1 : row half
    const int wc    = wid & 3;          // 0..3 : col quarter
    const int nBase = blockIdx.y * 64;
    const int cB    = blockIdx.x * 256;
    const int mrow  = lane & 15;        // A row / B column within 16x16 tile
    const int h     = lane >> 4;        // lane half selects K sub-range
    const float beta = beta_p[0];

    for (int i = threadIdx.x; i < 64 * NCLS; i += 256) (&lgt[0][0])[i] = 0.f;

    const unsigned ldsA_off = (unsigned)(uintptr_t)&ldsA[0][0][0];
    const unsigned ldsB_off = (unsigned)(uintptr_t)&ldsB[0][0][0];
    const bf16_t* gA = xs + (size_t)nBase * DDIM;
    const bf16_t* gB = eb + (size_t)cB * DDIM;

    // Prologue: stage chunk 0 (wave0 -> B tile, wave1 -> A tile).
    if (wid == 0)      tdm_load_2d(ldsB_off, gB, DDIM, MPAD, DDIM, KCHUNK, 256);
    else if (wid == 1) tdm_load_2d(ldsA_off, gA, DDIM, NDIM, DDIM, KCHUNK, 64);

    v8f acc[2][4] = {};

    for (int c = 0; c < NCHUNKS; ++c) {
        const int p = c & 1;
        // Issue next chunk into the other buffer, then wait for this chunk.
        if (wid == 0) {
            if (c + 1 < NCHUNKS) {
                tdm_load_2d(ldsB_off + (unsigned)(p ^ 1) * (256u * ROWB),
                            gB + (size_t)(c + 1) * KCHUNK, DDIM, MPAD, DDIM,
                            KCHUNK, 256);
                __builtin_amdgcn_s_wait_tensorcnt((short)1);
            } else {
                __builtin_amdgcn_s_wait_tensorcnt((short)0);
            }
        } else if (wid == 1) {
            if (c + 1 < NCHUNKS) {
                tdm_load_2d(ldsA_off + (unsigned)(p ^ 1) * (64u * ROWB),
                            gA + (size_t)(c + 1) * KCHUNK, DDIM, NDIM, DDIM,
                            KCHUNK, 64);
                __builtin_amdgcn_s_wait_tensorcnt((short)1);
            } else {
                __builtin_amdgcn_s_wait_tensorcnt((short)0);
            }
        }
        __syncthreads();   // chunk c resident in buffer p for all waves

        // A fragments (16x32 bf16): lane(m,h) reads 16B at bf16 offs 8h and 16+8h.
        Frag a0, a1;
        a0.q[0] = *(const uint4*)&ldsA[p][wr * 32 + mrow][16 * h];
        a0.q[1] = *(const uint4*)&ldsA[p][wr * 32 + mrow][32 + 16 * h];
        a1.q[0] = *(const uint4*)&ldsA[p][wr * 32 + 16 + mrow][16 * h];
        a1.q[1] = *(const uint4*)&ldsA[p][wr * 32 + 16 + mrow][32 + 16 * h];

        #pragma unroll
        for (int t = 0; t < 4; ++t) {
            // B fragment (32x16 bf16): lane(n,h) reads 16B at bf16 offs 8h and 16+8h.
            Frag b;
            b.q[0] = *(const uint4*)&ldsB[p][wc * 64 + t * 16 + mrow][16 * h];
            b.q[1] = *(const uint4*)&ldsB[p][wc * 64 + t * 16 + mrow][32 + 16 * h];
            acc[0][t] = __builtin_amdgcn_wmma_f32_16x16x32_bf16(
                false, a0.v, false, b.v, (short)0, acc[0][t], false, false);
            acc[1][t] = __builtin_amdgcn_wmma_f32_16x16x32_bf16(
                false, a1.v, false, b.v, (short)0, acc[1][t], false, false);
        }
        __syncthreads();   // everyone done with buffer p before it is re-filled
    }

    // C/D layout: VGPR v holds row 8h+v, column = lane&15.
    float xq[2][8];
    #pragma unroll
    for (int r = 0; r < 2; ++r)
        #pragma unroll
        for (int v = 0; v < 8; ++v)
            xq[r][v] = xsq[nBase + wr * 32 + r * 16 + 8 * h + v];

    #pragma unroll
    for (int r = 0; r < 2; ++r) {
        #pragma unroll
        for (int t = 0; t < 4; ++t) {
            const int mg  = cB + wc * 64 + t * 16 + mrow;
            const float es = esq[mg];
            const int lab = (mg < MDIM) ? labels[mg] : -1;
            #pragma unroll
            for (int v = 0; v < 8; ++v) {
                float dist = xq[r][v] + es - 2.f * acc[r][t][v];
                float att  = __expf(-beta * dist);
                if (lab >= 0) {
                    int lr = wr * 32 + r * 16 + 8 * h + v;
                    atomicAdd(&lgt[lr][lab], att);      // ds_add_f32
                }
            }
        }
    }

    __syncthreads();
    for (int i = threadIdx.x; i < 64 * NCLS; i += 256) {
        int lr = i / NCLS, cc = i % NCLS;
        atomicAdd(&logits[(size_t)(nBase + lr) * NCLS + cc], lgt[lr][cc]);
    }
}

// ---------------- softmax over 10 classes ----------------
__global__ void softmax_kernel(const float* __restrict__ logits,
                               const float* __restrict__ gamma_p,
                               float* __restrict__ out) {
    int n = blockIdx.x * blockDim.x + threadIdx.x;
    if (n >= NDIM) return;
    const float g = gamma_p[0];
    float v[NCLS];
    float mx = -INFINITY;
    #pragma unroll
    for (int c = 0; c < NCLS; ++c) {
        v[c] = g * logits[(size_t)n * NCLS + c];
        mx = fmaxf(mx, v[c]);
    }
    float s = 0.f;
    #pragma unroll
    for (int c = 0; c < NCLS; ++c) { v[c] = __expf(v[c] - mx); s += v[c]; }
    float inv = 1.f / s;
    #pragma unroll
    for (int c = 0; c < NCLS; ++c) out[(size_t)n * NCLS + c] = v[c] * inv;
}

// ---------------- launch ----------------
extern "C" void kernel_launch(void* const* d_in, const int* in_sizes, int n_in,
                              void* d_out, int out_size, void* d_ws, size_t ws_size,
                              hipStream_t stream) {
    const float* x      = (const float*)d_in[0];   // [1024, 512]
    const float* e      = (const float*)d_in[1];   // [50000, 512]
    const int*   labels = (const int*)d_in[2];     // [50000]
    const float* sig    = (const float*)d_in[3];   // [512]
    const float* beta   = (const float*)d_in[4];   // [1]
    const float* gamma  = (const float*)d_in[5];   // [1]
    float* out = (float*)d_out;                    // [1024, 10]

    char* ws = (char*)d_ws;
    size_t off = 0;
    auto carve = [&](size_t bytes) -> char* {
        char* p = ws + off;
        off = (off + bytes + 255) & ~(size_t)255;
        return p;
    };
    bf16_t* xs   = (bf16_t*)carve((size_t)NDIM * DDIM * sizeof(bf16_t)); // 1 MB
    bf16_t* ebuf = (bf16_t*)carve((size_t)MPAD * DDIM * sizeof(bf16_t)); // 49 MB
    float*  xsq  = (float*)carve((size_t)NDIM * sizeof(float));
    float*  esq  = (float*)carve((size_t)MPAD * sizeof(float));
    float*  lgt  = (float*)carve((size_t)NDIM * NCLS * sizeof(float));
    (void)ws_size; (void)in_sizes; (void)n_in; (void)out_size;

    prep_x_kernel<<<NDIM, 256, 0, stream>>>(x, sig, xs, xsq);
    prep_e_kernel<<<MPAD, 256, 0, stream>>>(e, sig, ebuf, esq);
    zero_kernel<<<(NDIM * NCLS + 255) / 256, 256, 0, stream>>>(lgt, NDIM * NCLS);

    dim3 grid(MPAD / 256, NDIM / 64);   // (196, 16)
    exemplar_main_kernel<<<grid, 256, 0, stream>>>(xs, ebuf, xsq, esq,
                                                   labels, beta, lgt);

    softmax_kernel<<<(NDIM + 255) / 256, 256, 0, stream>>>(lgt, gamma, out);
}
